// LSTM_Layer_80599356277103
// MI455X (gfx1250) — matrix operand order repaired and checked by reading, not compiled, not run on hardware
//
#include <hip/hip_runtime.h>

// ---------------------------------------------------------------------------
// LSTM layer on MI455X (gfx1250): bf16 WMMA (v_wmma_f32_16x16x32_bf16),
// one kernel launch per timestep (sequential recurrence), weights resident in
// L2 as pre-transposed bf16. K-loop split at the h/x boundary so all matrix
// operand loads are base + immediate-offset (no per-chunk cndmask selects).
// ---------------------------------------------------------------------------

#define BB 32      // batch
#define TT 512     // seq len
#define DD 512     // input dim
#define UU 1024    // units
#define KK 1536    // UU + DD (reduction dim)

typedef __attribute__((ext_vector_type(16))) __bf16 v16bf;
typedef __attribute__((ext_vector_type(8)))  float  v8f;

union BFrag {
    uint4  q[2];   // two b128 loads
    v16bf  v;      // WMMA operand
};

// ---------------------------------------------------------------------------
// Prep 1: transpose + convert each gate weight [K][U] f32 -> Wt[g][U][K] bf16
// (column-of-W becomes contiguous K run => B-fragment = two runs of 8 contiguous bf16).
// ---------------------------------------------------------------------------
__global__ __launch_bounds__(256) void prep_w_kernel(
    const float* __restrict__ Wf, const float* __restrict__ Wi,
    const float* __restrict__ Wc, const float* __restrict__ Wo,
    __bf16* __restrict__ Wt)
{
    __shared__ float tile[16][17];
    const int g  = blockIdx.z;
    const float* W = (g == 0) ? Wf : (g == 1) ? Wi : (g == 2) ? Wc : Wo;
    const int u0 = blockIdx.x * 16;
    const int k0 = blockIdx.y * 16;
    const int tx = threadIdx.x, ty = threadIdx.y;

    tile[ty][tx] = W[(size_t)(k0 + ty) * UU + (u0 + tx)];   // coalesced read
    __syncthreads();
    // Wt[g][u0+ty][k0+tx] = W[k0+tx][u0+ty]
    Wt[((size_t)g * UU + u0 + ty) * KK + k0 + tx] = (__bf16)tile[tx][ty];
}

// ---------------------------------------------------------------------------
// Prep 2: convert input activations f32 -> bf16 (same [B][T][D] layout).
// ---------------------------------------------------------------------------
__global__ __launch_bounds__(256) void prep_x_kernel(
    const float* __restrict__ x, __bf16* __restrict__ xbf, int n)
{
    int i = blockIdx.x * 256 + threadIdx.x;
    if (i < n) xbf[i] = (__bf16)x[i];
}

// ---------------------------------------------------------------------------
// Prep 3: zero-init h (both ping-pong bf16 buffers) and c (f32).
// ---------------------------------------------------------------------------
__global__ __launch_bounds__(256) void prep_state_kernel(
    __bf16* __restrict__ h0, __bf16* __restrict__ h1, float* __restrict__ c)
{
    int i = blockIdx.x * 256 + threadIdx.x;
    if (i < BB * UU) { h0[i] = (__bf16)0.f; h1[i] = (__bf16)0.f; c[i] = 0.f; }
}

__device__ __forceinline__ float sigmoidf_fast(float x)
{
    return 1.0f / (1.0f + __expf(-x));
}

// ---------------------------------------------------------------------------
// One LSTM timestep. Grid: UU/16 = 64 workgroups, 128 threads (4 waves).
// Wave w computes gate w's z[32 x 16] for this u-slice; K-loop split into the
// h-region (K 0..1023, from hin) and the x-region (K 1024..1535, from xbf) so
// every load is base-pointer + immediate offset. Gates meet in LDS, then
// pointwise c/h update for the owned u-slice.
// ---------------------------------------------------------------------------
__global__ __launch_bounds__(128) void lstm_step_kernel(
    const __bf16* __restrict__ Wt,    // [4][UU][KK] bf16
    const __bf16* __restrict__ xbf,   // [BB][TT][DD] bf16
    const float*  __restrict__ bf_,   // [UU]
    const float*  __restrict__ bi_,   // [UU]
    const float*  __restrict__ bc_,   // [UU]
    const float*  __restrict__ bo_,   // [UU]
    const __bf16* __restrict__ hin,   // [BB][UU] bf16 (read)
    __bf16*       __restrict__ hout,  // [BB][UU] bf16 (write)
    float*        __restrict__ cst,   // [BB][UU] f32 cell state
    float*        __restrict__ out,   // [BB][TT][UU] f32
    int t)
{
    __shared__ float zbuf[4][32][16];   // [gate][batch row][u within slice]

    const int tid  = threadIdx.x;
    const int wave = tid >> 5;          // gate id: 0=f 1=i 2=g 3=o
    const int lane = tid & 31;
    const int ln   = lane & 15;
    const int hi   = lane >> 4;         // lane half select
    const int u0   = blockIdx.x * 16;

    const int m0 = ln;                  // A-tile0 rows 0..15
    const int m1 = ln + 16;             // A-tile1 rows 16..31
    const int ka = hi ? 8 : 0;          // A K-chunk base within 32-wide step
    const int kb = hi ? 8 : 0;          // B K base within 32-wide step

    // Per-lane constant base pointers (all loop accesses: base + imm offset).
    const __bf16* a0h = hin + (size_t)m0 * UU + ka;
    const __bf16* a1h = hin + (size_t)m1 * UU + ka;
    const __bf16* a0x = xbf + ((size_t)m0 * TT + t) * DD + ka;
    const __bf16* a1x = xbf + ((size_t)m1 * TT + t) * DD + ka;
    const __bf16* bp  = Wt + ((size_t)(wave * UU + u0 + ln)) * KK + kb;

    v8f acc0 = {0.f, 0.f, 0.f, 0.f, 0.f, 0.f, 0.f, 0.f};
    v8f acc1 = {0.f, 0.f, 0.f, 0.f, 0.f, 0.f, 0.f, 0.f};

    // ---- Region 1: recurrent h contribution (K = 0..1023) ----
    #pragma unroll 4
    for (int k0 = 0; k0 < UU; k0 += 32) {
        BFrag a0, a1, b;
        a0.q[0] = *(const uint4*)(a0h + k0);
        a0.q[1] = *(const uint4*)(a0h + k0 + 16);
        a1.q[0] = *(const uint4*)(a1h + k0);
        a1.q[1] = *(const uint4*)(a1h + k0 + 16);
        b.q[0]  = *(const uint4*)(bp + k0);
        b.q[1]  = *(const uint4*)(bp + k0 + 16);

        acc0 = __builtin_amdgcn_wmma_f32_16x16x32_bf16(
                   false, a0.v, false, b.v, (short)0, acc0, false, false);
        acc1 = __builtin_amdgcn_wmma_f32_16x16x32_bf16(
                   false, a1.v, false, b.v, (short)0, acc1, false, false);
    }

    // ---- Region 2: input x_t contribution (K = 1024..1535) ----
    const __bf16* bp2 = bp + UU;
    #pragma unroll 4
    for (int k0 = 0; k0 < DD; k0 += 32) {
        BFrag a0, a1, b;
        a0.q[0] = *(const uint4*)(a0x + k0);
        a0.q[1] = *(const uint4*)(a0x + k0 + 16);
        a1.q[0] = *(const uint4*)(a1x + k0);
        a1.q[1] = *(const uint4*)(a1x + k0 + 16);
        b.q[0]  = *(const uint4*)(bp2 + k0);
        b.q[1]  = *(const uint4*)(bp2 + k0 + 16);

        acc0 = __builtin_amdgcn_wmma_f32_16x16x32_bf16(
                   false, a0.v, false, b.v, (short)0, acc0, false, false);
        acc1 = __builtin_amdgcn_wmma_f32_16x16x32_bf16(
                   false, a1.v, false, b.v, (short)0, acc1, false, false);
    }

    // Spill z tiles to LDS. C/D layout: VGPR r, lanes 0-15: M=r,N=lane;
    // lanes 16-31: M=r+8,N=lane-16.
    {
        const int mbase = hi ? 8 : 0;
        #pragma unroll
        for (int r = 0; r < 8; ++r) {
            zbuf[wave][mbase + r][ln]      = acc0[r];
            zbuf[wave][mbase + r + 16][ln] = acc1[r];
        }
    }
    __syncthreads();

    // Pointwise gates + state update for this u-slice (512 elems / 128 thr).
    #pragma unroll
    for (int idx = tid; idx < BB * 16; idx += 128) {
        const int m = idx >> 4;
        const int n = idx & 15;
        const int u = u0 + n;

        const float zf = zbuf[0][m][n] + bf_[u];
        const float zi = zbuf[1][m][n] + bi_[u];
        const float zg = zbuf[2][m][n] + bc_[u];
        const float zo = zbuf[3][m][n] + bo_[u];

        const float f = sigmoidf_fast(zf);
        const float i = sigmoidf_fast(zi);
        const float g = tanhf(zg);
        const float o = sigmoidf_fast(zo);

        const size_t cidx = (size_t)m * UU + u;
        const float cv = f * cst[cidx] + i * g;
        cst[cidx] = cv;
        const float h = o * tanhf(cv);

        out[((size_t)m * TT + t) * UU + u] = h;
        hout[cidx] = (__bf16)h;
    }
}

// ---------------------------------------------------------------------------
// Host launcher
// ---------------------------------------------------------------------------
extern "C" void kernel_launch(void* const* d_in, const int* in_sizes, int n_in,
                              void* d_out, int out_size, void* d_ws, size_t ws_size,
                              hipStream_t stream)
{
    (void)in_sizes; (void)n_in; (void)out_size; (void)ws_size;

    const float* data = (const float*)d_in[0];
    const float* Wf   = (const float*)d_in[1];
    const float* bfb  = (const float*)d_in[2];
    const float* Wi   = (const float*)d_in[3];
    const float* bib  = (const float*)d_in[4];
    const float* Wc   = (const float*)d_in[5];
    const float* bcb  = (const float*)d_in[6];
    const float* Wo   = (const float*)d_in[7];
    const float* bob  = (const float*)d_in[8];
    float* out        = (float*)d_out;

    // Workspace carve-up (256B aligned slices).
    char*  base = (char*)d_ws;
    size_t off  = 0;
    auto carve = [&](size_t bytes) -> char* {
        char* p = base + off;
        off = (off + bytes + 255) & ~(size_t)255;
        return p;
    };
    __bf16* Wt  = (__bf16*)carve((size_t)4 * UU * KK * sizeof(__bf16));  // 12.6MB
    __bf16* xbf = (__bf16*)carve((size_t)BB * TT * DD * sizeof(__bf16)); // 16.8MB
    __bf16* h0  = (__bf16*)carve((size_t)BB * UU * sizeof(__bf16));
    __bf16* h1  = (__bf16*)carve((size_t)BB * UU * sizeof(__bf16));
    float*  cst = (float*) carve((size_t)BB * UU * sizeof(float));

    // --- prep (every call: deterministic, no cached state) ---
    prep_w_kernel<<<dim3(UU / 16, KK / 16, 4), dim3(16, 16), 0, stream>>>(
        Wf, Wi, Wc, Wo, Wt);

    const int nx = BB * TT * DD;
    prep_x_kernel<<<(nx + 255) / 256, 256, 0, stream>>>(data, xbf, nx);

    prep_state_kernel<<<(BB * UU + 255) / 256, 256, 0, stream>>>(h0, h1, cst);

    // --- sequential recurrence: one kernel per timestep ---
    for (int t = 0; t < TT; ++t) {
        const __bf16* hin  = (t & 1) ? h1 : h0;
        __bf16*       hout = (t & 1) ? h0 : h1;
        lstm_step_kernel<<<UU / 16, 128, 0, stream>>>(
            Wt, xbf, bfb, bib, bcb, bob, hin, hout, cst, out, t);
    }
}
